// CGGRU_Interactions_78073915507064
// MI455X (gfx1250) — matrix operand order repaired and checked by reading, not compiled, not run on hardware
//
#include <hip/hip_runtime.h>
#include <hip/hip_fp16.h>

// MI455X / gfx1250, wave32. Core math path uses v_wmma_f32_16x16x32_f16.
//
// Fused NNConv message kernel: msg = P @ Q with P[e,(i,h)] = x_e[i]*hid_e[h]
// built in registers as WMMA A fragments (K=4096 streamed), Q = 512KB
// pre-swizzled f16 B-fragments (L2 resident). Avoids materializing the
// 1.6GB per-edge weight tensor entirely (4.8GB of HBM traffic saved).
// Each wave owns TWO 16-edge M-tiles so every B fragment feeds two WMMAs
// (halves L2 fragment traffic to ~1.6GB/pass).

typedef __attribute__((ext_vector_type(16))) _Float16 v16h;
typedef __attribute__((ext_vector_type(8)))  float    v8f;

#define HC 64
#define NF 64
#define NG 5

static __device__ __forceinline__ v8f wmma16(v16h a, v16h b, v8f c) {
  // D = A(16x32 f16) * B(32x16 f16) + C(16x16 f32)
  return __builtin_amdgcn_wmma_f32_16x16x32_f16(
      /*neg_a=*/false, a, /*neg_b=*/false, b,
      /*c_mod=*/(short)0, c, /*reuse_a=*/false, /*reuse_b=*/false);
}

// ---------------------------------------------------------------- utilities
__global__ void k_zero(float* p, long n) {
  long i = (long)blockIdx.x * blockDim.x + threadIdx.x;
  if (i < n) p[i] = 0.f;
}

__global__ void k_cnt(const int* __restrict__ dst, float* cnt, int E) {
  int e = blockIdx.x * blockDim.x + threadIdx.x;
  if (e < E) atomicAdd(&cnt[dst[e]], 1.0f);
}

// ------------------------------------------------- weight swizzle (generic)
// Pack B[K x NC] (f32) into WMMA-B fragment order (f16):
//   frag f = kc*nct + ct ; per-lane 16 f16 contiguous at (f*32 + lane)*16
//   element j of lane l:  k = kc*32 + (j>>3)*16 + (l>>4)*8 + (j&7),  c = ct*16 + (l&15)
__global__ void k_swzB(const float* __restrict__ src, _Float16* __restrict__ dstp,
                       int K, int NC, int transpose) {
  int idx = blockIdx.x * blockDim.x + threadIdx.x;
  if (idx >= K * NC) return;
  int j = idx & 15;
  int l = (idx >> 4) & 31;
  int f = idx >> 9;
  int nct = NC >> 4;
  int kc = f / nct, ct = f % nct;
  int k = kc * 32 + ((j >> 3) << 4) + ((l >> 4) << 3) + (j & 7);
  int c = ct * 16 + (l & 15);
  float v = transpose ? src[(size_t)c * K + k] : src[(size_t)k * NC + c];
  dstp[idx] = (_Float16)v;
}

// Pack Q[k=(i*64+h), o] = nn2_w[(i*64+o)*64 + h] into fragment order, K=4096, NC=64
__global__ void k_swzM2(const float* __restrict__ nn2w, _Float16* __restrict__ dstp) {
  int idx = blockIdx.x * blockDim.x + threadIdx.x;  // 4096*64 elements
  if (idx >= 4096 * 64) return;
  int j = idx & 15;
  int l = (idx >> 4) & 31;
  int f = idx >> 9;            // nct = 4
  int kc = f >> 2, ct = f & 3;
  int k = kc * 32 + ((j >> 3) << 4) + ((l >> 4) << 3) + (j & 7);
  int o = ct * 16 + (l & 15);
  int i = k >> 6, h = k & 63;
  dstp[idx] = (_Float16)nn2w[((size_t)(i * 64 + o)) * 64 + h];
}

// ------------------------------------------------------------- edge MLP
// hid[e,h] = relu( nn1( relu( edge_attr[e]@short_w.T + short_b ) ) )  -> f16
__global__ void k_edge_mlp(const float* __restrict__ eattr,
                           const float* __restrict__ sw, const float* __restrict__ sb,
                           const float* __restrict__ n1w, const float* __restrict__ n1b,
                           _Float16* __restrict__ hid, int E) {
  long idx = (long)blockIdx.x * blockDim.x + threadIdx.x;
  if (idx >= (long)E * 64) return;
  int e = (int)(idx >> 6), h = (int)(idx & 63);
  const float* a = eattr + (size_t)e * NG;
  float ea[3];
#pragma unroll
  for (int j = 0; j < 3; j++) {
    float s = sb[j];
#pragma unroll
    for (int g = 0; g < NG; g++) s += a[g] * sw[j * NG + g];
    ea[j] = fmaxf(s, 0.f);
  }
  float t = n1b[h] + ea[0] * n1w[h * 3 + 0] + ea[1] * n1w[h * 3 + 1] + ea[2] * n1w[h * 3 + 2];
  hid[idx] = (_Float16)fmaxf(t, 0.f);
}

// --------------------------------------------- generic WMMA GEMM, K = 64
// Y[R,NC] = RELU?( X[R,64] @ Bfrags + bias ) ; one wave per 16-row tile.
// NC/RELU are template params: fully unrolled column loop, no flag branches.
template <int NC, bool RELU>
__global__ void __launch_bounds__(256)
k_gemm64(const float* __restrict__ X, const _Float16* __restrict__ Bf,
         const float* __restrict__ bias, float* __restrict__ Y, int R) {
  int wave = blockIdx.x * (blockDim.x >> 5) + (threadIdx.x >> 5);
  if (wave >= (R >> 4)) return;
  int l = threadIdx.x & 31, m = l & 15, kh = l >> 4;
  int row0 = wave << 4;

  // Build two A fragments (f32 -> f16). A element j: kk = kh*8 + (j&7) + (j>>3)*16
  const float* xr = X + (size_t)(row0 + m) * 64;
  v16h A[2];
#pragma unroll
  for (int kc = 0; kc < 2; kc++) {
    const float4* p1 = (const float4*)(xr + kc * 32 + kh * 8);
    const float4* p2 = (const float4*)(xr + kc * 32 + 16 + kh * 8);
    float4 r0 = p1[0], r1 = p1[1], r2 = p2[0], r3 = p2[1];
    A[kc][0] = (_Float16)r0.x;  A[kc][1] = (_Float16)r0.y;
    A[kc][2] = (_Float16)r0.z;  A[kc][3] = (_Float16)r0.w;
    A[kc][4] = (_Float16)r1.x;  A[kc][5] = (_Float16)r1.y;
    A[kc][6] = (_Float16)r1.z;  A[kc][7] = (_Float16)r1.w;
    A[kc][8] = (_Float16)r2.x;  A[kc][9] = (_Float16)r2.y;
    A[kc][10] = (_Float16)r2.z; A[kc][11] = (_Float16)r2.w;
    A[kc][12] = (_Float16)r3.x; A[kc][13] = (_Float16)r3.y;
    A[kc][14] = (_Float16)r3.z; A[kc][15] = (_Float16)r3.w;
  }
  constexpr int nct = NC >> 4;
#pragma unroll
  for (int ct = 0; ct < nct; ct++) {
    v16h b0 = *(const v16h*)(Bf + ((size_t)(0 * nct + ct) * 32 + l) * 16);
    v16h b1 = *(const v16h*)(Bf + ((size_t)(1 * nct + ct) * 32 + l) * 16);
    v8f c = {};
    c = wmma16(A[0], b0, c);
    c = wmma16(A[1], b1, c);
    int col = ct * 16 + m;
    float bv = bias[col];
#pragma unroll
    for (int r = 0; r < 8; r++) {
      float v = c[r] + bv;
      if (RELU) v = fmaxf(v, 0.f);
      Y[(size_t)(row0 + r + kh * 8) * NC + col] = v;
    }
  }
}

// --------------------------------- fused NNConv message kernel (the beast)
// Per wave: 32 edges as two 16-row M-tiles sharing every B fragment.
// msg[32,64] = P[32,4096] @ Q[4096,64] + X@NB, then float-atomic scatter-add.
__global__ void __launch_bounds__(256)
k_msg(const float* __restrict__ X, const _Float16* __restrict__ hid,
      const _Float16* __restrict__ M2B, const _Float16* __restrict__ NBB,
      const int* __restrict__ src, const int* __restrict__ dst,
      float* __restrict__ agg, int E) {
  __shared__ _Float16 xsh[8][2048];  // 32 rows x 64, per wave
  __shared__ _Float16 hsh[8][2048];
  __shared__ int      dsh[8][32];

  int w = threadIdx.x >> 5, l = threadIdx.x & 31, m = l & 15, kh = l >> 4;
  int grp = blockIdx.x * 8 + w;     // 32-edge group
  int ngrp = E >> 5;
  bool active = grp < ngrp;

  if (active) {
    int e = (grp << 5) + l;         // each lane stages one full edge row
    int s = src[e];
    dsh[w][l] = dst[e];
    const float4* xp = (const float4*)(X + (size_t)s * 64);
    _Float16* xq = &xsh[w][l * 64];
#pragma unroll
    for (int q = 0; q < 16; q++) {
      float4 v = xp[q];
      xq[q * 4 + 0] = (_Float16)v.x; xq[q * 4 + 1] = (_Float16)v.y;
      xq[q * 4 + 2] = (_Float16)v.z; xq[q * 4 + 3] = (_Float16)v.w;
    }
    const uint4* hp = (const uint4*)(hid + (size_t)e * 64);
    uint4* hq = (uint4*)&hsh[w][l * 64];
    hq[0] = hp[0]; hq[1] = hp[1]; hq[2] = hp[2]; hq[3] = hp[3];
  }
  __syncthreads();
  if (!active) return;

  // Hidden values pre-arranged in A-fragment element order, per M-tile t.
  v16h hidA[2][2];
#pragma unroll
  for (int t = 0; t < 2; t++)
#pragma unroll
    for (int p = 0; p < 2; p++) {
      int base = (t * 16 + m) * 64 + p * 32 + kh * 8;
#pragma unroll
      for (int j = 0; j < 8; j++) {
        hidA[t][p][j]     = hsh[w][base + j];
        hidA[t][p][8 + j] = hsh[w][base + 16 + j];
      }
    }

  // Seed accumulators with bias GEMM: C[t] = Xtile[t] @ NB[64,64]
  v8f c[2][4];
  {
    v16h ab[2][2];
#pragma unroll
    for (int t = 0; t < 2; t++)
#pragma unroll
      for (int kc = 0; kc < 2; kc++) {
        int base = (t * 16 + m) * 64 + kc * 32 + kh * 8;
#pragma unroll
        for (int j = 0; j < 8; j++) {
          ab[t][kc][j]     = xsh[w][base + j];
          ab[t][kc][8 + j] = xsh[w][base + 16 + j];
        }
      }
#pragma unroll
    for (int ct = 0; ct < 4; ct++) {
      v16h b0 = *(const v16h*)(NBB + ((size_t)(0 * 4 + ct) * 32 + l) * 16);
      v16h b1 = *(const v16h*)(NBB + ((size_t)(1 * 4 + ct) * 32 + l) * 16);
#pragma unroll
      for (int t = 0; t < 2; t++) {
        v8f z = {};
        z = wmma16(ab[t][0], b0, z);
        c[t][ct] = wmma16(ab[t][1], b1, z);
      }
    }
  }

  // Main contraction: K = 4096 = 128 chunks of 32; chunk kc has i = kc>>1
  // constant, h = (kc&1)*32 + kk. A[t] = x[t][m][i] * hidA[t][kc&1]
  // (v_pk_mul_f16 via ext-vector * scalar splat). Each B fragment feeds
  // both M-tiles.
  const v16h* m2 = (const v16h*)M2B + l;   // lane base, v16h = 32B chunks
  for (int kc = 0; kc < 128; kc++) {
    _Float16 xv0 = xsh[w][m * 64 + (kc >> 1)];
    _Float16 xv1 = xsh[w][(16 + m) * 64 + (kc >> 1)];
    v16h a0 = hidA[0][kc & 1] * xv0;
    v16h a1 = hidA[1][kc & 1] * xv1;
    int pk = kc + 8 < 128 ? kc + 8 : 127;       // stream prefetch ~32KB ahead
    __builtin_prefetch((const void*)(m2 + (size_t)pk * 128), 0, 1);
#pragma unroll
    for (int ct = 0; ct < 4; ct++) {
      v16h b = m2[(size_t)(kc * 4 + ct) * 32];
      c[0][ct] = wmma16(a0, b, c[0][ct]);
      c[1][ct] = wmma16(a1, b, c[1][ct]);
    }
  }

  // Scatter-add per (edge, o) into agg[dst]
#pragma unroll
  for (int t = 0; t < 2; t++)
#pragma unroll
    for (int ct = 0; ct < 4; ct++) {
      int col = ct * 16 + m;
#pragma unroll
      for (int r = 0; r < 8; r++) {
        int d = dsh[w][t * 16 + r + kh * 8];
        atomicAdd(&agg[(size_t)d * 64 + col], c[t][ct][r]);
      }
    }
}

// --------------------------------------------------------- node elementwise
__global__ void k_node_m(const float* __restrict__ agg, const float* __restrict__ cnt,
                         const float* __restrict__ rootmul, float* __restrict__ m,
                         int N) {
  long idx = (long)blockIdx.x * blockDim.x + threadIdx.x;
  if (idx >= (long)N * 64) return;
  int n = (int)(idx >> 6);
  float inv = 1.0f / fmaxf(cnt[n], 1.0f);
  m[idx] = fmaxf(agg[idx] * inv + rootmul[idx], 0.f);  // rootmul carries conv_b
}

__global__ void k_gru(const float* __restrict__ gi, const float* __restrict__ gh,
                      const float* __restrict__ hprev, float* __restrict__ hout,
                      int N) {
  long idx = (long)blockIdx.x * blockDim.x + threadIdx.x;
  if (idx >= (long)N * 64) return;
  int n = (int)(idx >> 6), cc = (int)(idx & 63);
  size_t b = (size_t)n * 192;
  float r = 1.f / (1.f + __expf(-(gi[b + cc] + gh[b + cc])));
  float z = 1.f / (1.f + __expf(-(gi[b + 64 + cc] + gh[b + 64 + cc])));
  float nn = tanhf(gi[b + 128 + cc] + r * gh[b + 128 + cc]);
  hout[idx] = (1.f - z) * nn + z * hprev[idx];
}

// ------------------------------------------------------------------ launch
extern "C" void kernel_launch(void* const* d_in, const int* in_sizes, int n_in,
                              void* d_out, int out_size, void* d_ws, size_t ws_size,
                              hipStream_t stream) {
  const float* h        = (const float*)d_in[0];
  const int*   ei       = (const int*)d_in[1];
  const float* eattr    = (const float*)d_in[3];
  const float* lin0_w   = (const float*)d_in[4];
  const float* lin0_b   = (const float*)d_in[5];
  const float* short_w  = (const float*)d_in[6];
  const float* short_b  = (const float*)d_in[7];
  const float* nn1_w    = (const float*)d_in[8];
  const float* nn1_b    = (const float*)d_in[9];
  const float* nn2_w    = (const float*)d_in[10];
  const float* nn2_b    = (const float*)d_in[11];
  const float* root_w   = (const float*)d_in[12];
  const float* conv_b   = (const float*)d_in[13];
  const float* gru_wih  = (const float*)d_in[14];
  const float* gru_whh  = (const float*)d_in[15];
  const float* gru_bih  = (const float*)d_in[16];
  const float* gru_bhh  = (const float*)d_in[17];

  const int N = in_sizes[0] / HC;
  const int E = in_sizes[2];
  const int* src = ei;
  const int* dst = ei + E;
  float* hstate = (float*)d_out;  // hstate lives in d_out

  // Workspace carve-up (~270 MB)
  char* ws = (char*)d_ws;
  size_t off = 0;
  auto carve = [&](size_t bytes) -> char* {
    char* p = ws + off;
    off = (off + bytes + 255) & ~(size_t)255;
    return p;
  };
  _Float16* hiddenF16 = (_Float16*)carve((size_t)E * 64 * 2);
  _Float16* M2B   = (_Float16*)carve((size_t)4096 * 64 * 2);
  _Float16* NBB   = (_Float16*)carve((size_t)64 * 64 * 2);
  _Float16* lin0B = (_Float16*)carve((size_t)64 * 64 * 2);
  _Float16* rootB = (_Float16*)carve((size_t)64 * 64 * 2);
  _Float16* wihB  = (_Float16*)carve((size_t)64 * 192 * 2);
  _Float16* whhB  = (_Float16*)carve((size_t)64 * 192 * 2);
  float* out0    = (float*)carve((size_t)N * 64 * 4);
  float* agg     = (float*)carve((size_t)N * 64 * 4);
  float* rootmul = (float*)carve((size_t)N * 64 * 4);
  float* mbuf    = (float*)carve((size_t)N * 64 * 4);
  float* gi      = (float*)carve((size_t)N * 192 * 4);
  float* gh      = (float*)carve((size_t)N * 192 * 4);
  float* cnt     = (float*)carve((size_t)N * 4);
  (void)ws_size;

  const int B = 256;
  auto cdiv = [](long a, long b) { return (int)((a + b - 1) / b); };

  // --- prep: weight swizzles into WMMA B-fragment order
  k_swzB<<<cdiv(64 * 64, B), B, 0, stream>>>(lin0_w, lin0B, 64, 64, 1);   // h @ lin0_w.T
  k_swzB<<<cdiv(64 * 64, B), B, 0, stream>>>(root_w, rootB, 64, 64, 0);   // out @ root_w
  k_swzB<<<cdiv(64 * 192, B), B, 0, stream>>>(gru_wih, wihB, 64, 192, 1); // m @ wih.T
  k_swzB<<<cdiv(64 * 192, B), B, 0, stream>>>(gru_whh, whhB, 64, 192, 1); // h @ whh.T
  k_swzB<<<cdiv(64 * 64, B), B, 0, stream>>>(nn2_b, NBB, 64, 64, 0);      // NB[i,o]
  k_swzM2<<<cdiv(4096 * 64, B), B, 0, stream>>>(nn2_w, M2B);

  // --- edge MLP (hidden, f16) and in-degree counts
  k_edge_mlp<<<cdiv((long)E * 64, B), B, 0, stream>>>(eattr, short_w, short_b,
                                                      nn1_w, nn1_b, hiddenF16, E);
  k_zero<<<cdiv(N, B), B, 0, stream>>>(cnt, N);
  k_cnt<<<cdiv(E, B), B, 0, stream>>>(dst, cnt, E);

  // --- lin0: out0 = relu(h @ lin0_w.T + lin0_b)   [WMMA]
  int gemmBlocks = cdiv(N >> 4, 8);
  k_gemm64<64, true><<<gemmBlocks, B, 0, stream>>>(h, lin0B, lin0_b, out0, N);

  int msgBlocks = cdiv(E >> 5, 8);
  for (int it = 0; it < 2; it++) {
    const float* cur = (it == 0) ? out0 : hstate;  // out == hstate always

    k_zero<<<cdiv((long)N * 64, B), B, 0, stream>>>(agg, (long)N * 64);
    k_msg<<<msgBlocks, B, 0, stream>>>(cur, hiddenF16, M2B, NBB, src, dst, agg, E);
    // rootmul = cur @ root_w + conv_b   [WMMA]
    k_gemm64<64, false><<<gemmBlocks, B, 0, stream>>>(cur, rootB, conv_b, rootmul, N);
    k_node_m<<<cdiv((long)N * 64, B), B, 0, stream>>>(agg, cnt, rootmul, mbuf, N);
    // gi = m @ wih.T + bih ; gh = hstate @ whh.T + bhh   [WMMA]
    k_gemm64<192, false><<<gemmBlocks, B, 0, stream>>>(mbuf, wihB, gru_bih, gi, N);
    k_gemm64<192, false><<<gemmBlocks, B, 0, stream>>>(cur, whhB, gru_bhh, gh, N);
    k_gru<<<cdiv((long)N * 64, B), B, 0, stream>>>(gi, gh, cur, hstate, N);
  }
}
